// GATNet_79216376808034
// MI455X (gfx1250) — matrix operand order repaired and checked by reading, not compiled, not run on hardware
//
#include <hip/hip_runtime.h>
#include <hip/hip_bf16.h>
#include <math.h>

// ---------------- problem constants (match reference) ----------------
#define NN     10000
#define EE     160000
#define GG     64
#define EDIMC  32
#define HHEADS 4
#define E2TOT  (EE + NN)          // edges + self loops
#define NEG_SLOPE 0.2f

typedef __attribute__((ext_vector_type(16))) __bf16 v16bf;
typedef __attribute__((ext_vector_type(8)))  float  v8f;

// ---------------- small device helpers ----------------
__device__ __forceinline__ void edge_sd(const int* __restrict__ ei, int e, int& s, int& d) {
    if (e < EE) { s = ei[e]; d = ei[EE + e]; }
    else        { s = d = e - EE; }          // appended self loop
}

__device__ __forceinline__ void atomicMaxF(float* addr, float v) {
    // signed-int max for positives, unsigned-int min for negatives.
    if (v >= 0.f) atomicMax((int*)addr, __float_as_int(v));
    else          atomicMin((unsigned int*)addr, (unsigned int)__float_as_int(v));
}

__global__ void fill_f32(float* p, float v, int n) {
    int i = blockIdx.x * blockDim.x + threadIdx.x;
    if (i < n) p[i] = v;
}

// ---------------- self-loop edge_attr = scatter-mean ----------------
__global__ void selfloop_accum(const int* __restrict__ ei, const float* __restrict__ eattr,
                               float* __restrict__ meanE, float* __restrict__ deg) {
    int e = blockIdx.x * blockDim.x + threadIdx.x;
    if (e >= EE) return;
    int d = ei[EE + e];
    atomicAdd(&deg[d], 1.0f);
    const float* r = eattr + (size_t)e * EDIMC;
    float* o = meanE + (size_t)d * EDIMC;
    #pragma unroll
    for (int k = 0; k < EDIMC; ++k) atomicAdd(&o[k], r[k]);
}

__global__ void selfloop_div(float* __restrict__ meanE, const float* __restrict__ deg) {
    int i = blockIdx.x * blockDim.x + threadIdx.x;
    if (i >= NN * EDIMC) return;
    meanE[i] /= fmaxf(deg[i / EDIMC], 1.0f);
}

// ---------------- bf16 WMMA GEMM: C[M,Nc] = A[M,K] @ B[K,Nc] (fp32 io) ----------------
// Requires: M % 16 == 0 (true here: M = 10000), K % 32 == 0, Nc % 64 == 0.
// One wave computes a 16x64 strip: one A fragment reused across 4 B fragments / 4 WMMAs
// per K-step of 32.
// A-frag lane map (16-bit A 16x32, contiguous k): lane l: m=l&15, kh=l>>4;
//   elem j(0..15) -> K = k0+kh*16+j (same k order as B)
// B-frag lane map (KxN): lane l: n=l&15, kh=l>>4; elem j(0..15) -> K = k0+kh*16+j
// C/D layout: VGPR r, lanes 0-15 -> (M=r, N=lane); lanes 16-31 -> (M=8+r, N=lane-16)
__global__ void gemm_bf16_wmma(const float* __restrict__ A, const float* __restrict__ B,
                               float* __restrict__ C, int M, int K, int Nc) {
    int wave = blockIdx.x * (blockDim.x >> 5) + (threadIdx.x >> 5);
    int lane = threadIdx.x & 31;
    int ntn4 = Nc >> 6;                         // tiles of 64 columns
    int ntm  = M >> 4;
    if (wave >= ntm * ntn4) return;             // wave-uniform: EXEC all-ones below
    int tm = wave / ntn4, tn4 = wave % ntn4;
    int ml = lane & 15;
    int kh = lane >> 4;
    const float* Arow = A + (size_t)(tm * 16 + ml) * K;
    int col0 = tn4 * 64 + ml;

    v8f acc0 = {}, acc1 = {}, acc2 = {}, acc3 = {};
    for (int k0 = 0; k0 < K; k0 += 32) {
        // prefetch next K-step (emits global_prefetch_b8)
        if (k0 + 32 < K) {
            __builtin_prefetch(Arow + k0 + 32 + kh * 8, 0, 3);
            __builtin_prefetch(B + (size_t)(k0 + 32 + kh * 16) * Nc + col0, 0, 3);
        }
        // ---- A fragment: 4x float4 vector loads, packed cvt to bf16 ----
        const float4* Ap4 = (const float4*)(Arow + k0 + kh * 16);
        float4 f0 = Ap4[0];           // K = k0+kh*16    .. +3
        float4 f1 = Ap4[1];           // K = k0+kh*16+4  .. +7
        float4 f2 = Ap4[2];           // K = k0+kh*16+8  .. +11
        float4 f3 = Ap4[3];           // K = k0+kh*16+12 .. +15
        v16bf a;
        a[0]  = (__bf16)f0.x; a[1]  = (__bf16)f0.y; a[2]  = (__bf16)f0.z; a[3]  = (__bf16)f0.w;
        a[4]  = (__bf16)f1.x; a[5]  = (__bf16)f1.y; a[6]  = (__bf16)f1.z; a[7]  = (__bf16)f1.w;
        a[8]  = (__bf16)f2.x; a[9]  = (__bf16)f2.y; a[10] = (__bf16)f2.z; a[11] = (__bf16)f2.w;
        a[12] = (__bf16)f3.x; a[13] = (__bf16)f3.y; a[14] = (__bf16)f3.z; a[15] = (__bf16)f3.w;

        // ---- 4 B fragments (lane-coalesced rows of B) + 4 WMMAs ----
        const float* Bbase = B + (size_t)(k0 + kh * 16) * Nc + col0;
        v16bf b0, b1, b2, b3;
        #pragma unroll
        for (int j = 0; j < 16; ++j) {
            const float* Br = Bbase + (size_t)j * Nc;
            b0[j] = (__bf16)Br[0];
            b1[j] = (__bf16)Br[16];
            b2[j] = (__bf16)Br[32];
            b3[j] = (__bf16)Br[48];
        }
        acc0 = __builtin_amdgcn_wmma_f32_16x16x32_bf16(false, a, false, b0, (short)0, acc0, false, false);
        acc1 = __builtin_amdgcn_wmma_f32_16x16x32_bf16(false, a, false, b1, (short)0, acc1, false, false);
        acc2 = __builtin_amdgcn_wmma_f32_16x16x32_bf16(false, a, false, b2, (short)0, acc2, false, false);
        acc3 = __builtin_amdgcn_wmma_f32_16x16x32_bf16(false, a, false, b3, (short)0, acc3, false, false);
    }

    // ---- store: rows tm*16 + kh*8 + r, cols tn4*64 + t*16 + ml ----
    float* Cbase = C + (size_t)(tm * 16 + kh * 8) * Nc + tn4 * 64 + ml;
    #pragma unroll
    for (int r = 0; r < 8; ++r) {
        float* Cr = Cbase + (size_t)r * Nc;
        Cr[0]  = acc0[r];
        Cr[16] = acc1[r];
        Cr[32] = acc2[r];
        Cr[48] = acc3[r];
    }
}

// ---------------- fold We @ a_e -> wevec[32,H] (kills the edge GEMM) ----------------
__global__ void fold_wevec(const float* __restrict__ We, const float* __restrict__ ae,
                           float* __restrict__ wevec, int cout) {
    int i = blockIdx.x * blockDim.x + threadIdx.x;
    if (i >= EDIMC * HHEADS) return;
    int k = i / HHEADS, h = i % HHEADS;
    int HC = HHEADS * cout;
    float acc = 0.f;
    for (int c = 0; c < cout; ++c) acc += We[(size_t)k * HC + h * cout + c] * ae[h * cout + c];
    wevec[k * HHEADS + h] = acc;
}

// ---------------- per-node attention scalars from xs ----------------
__global__ void node_att(const float* __restrict__ xs, const float* __restrict__ as_,
                         const float* __restrict__ ad_, float* __restrict__ alsrc,
                         float* __restrict__ aldst, int cout) {
    int i = blockIdx.x * blockDim.x + threadIdx.x;
    if (i >= NN * HHEADS) return;
    int n = i / HHEADS, h = i % HHEADS;
    const float* xr = xs + (size_t)n * HHEADS * cout + h * cout;
    float s = 0.f, d = 0.f;
    for (int c = 0; c < cout; ++c) { float v = xr[c]; s += v * as_[h * cout + c]; d += v * ad_[h * cout + c]; }
    alsrc[i] = s; aldst[i] = d;
}

// ---------------- softmax pass 1: logits + per-dst max ----------------
__global__ void att_logit_max(const int* __restrict__ ei, const float* __restrict__ eattr,
                              const float* __restrict__ meanE, const float* __restrict__ wevec,
                              const float* __restrict__ alsrc, const float* __restrict__ aldst,
                              float* __restrict__ logits, float* __restrict__ mbuf) {
    int i = blockIdx.x * blockDim.x + threadIdx.x;
    if (i >= E2TOT * HHEADS) return;
    int e = i / HHEADS, h = i % HHEADS;
    int s, d; edge_sd(ei, e, s, d);
    const float* er = (e < EE) ? (eattr + (size_t)e * EDIMC) : (meanE + (size_t)(e - EE) * EDIMC);
    float ale = 0.f;
    #pragma unroll
    for (int k = 0; k < EDIMC; ++k) ale += er[k] * wevec[k * HHEADS + h];
    float lg = alsrc[s * HHEADS + h] + aldst[d * HHEADS + h] + ale;
    lg = (lg > 0.f) ? lg : NEG_SLOPE * lg;          // leaky_relu
    logits[i] = lg;
    atomicMaxF(&mbuf[d * HHEADS + h], lg);
}

// ---------------- softmax pass 2: exp + per-dst sum ----------------
__global__ void att_exp_sum(const int* __restrict__ ei, float* __restrict__ logits,
                            const float* __restrict__ mbuf, float* __restrict__ sbuf) {
    int i = blockIdx.x * blockDim.x + threadIdx.x;
    if (i >= E2TOT * HHEADS) return;
    int e = i / HHEADS, h = i % HHEADS;
    int s, d; edge_sd(ei, e, s, d); (void)s;
    float ex = __expf(logits[i] - mbuf[d * HHEADS + h]);
    logits[i] = ex;
    atomicAdd(&sbuf[d * HHEADS + h], ex);
}

// ---------------- weighted scatter: out[dst] += alpha * xs[src] ----------------
__global__ void aggregate(const int* __restrict__ ei, const float* __restrict__ exbuf,
                          const float* __restrict__ sbuf, const float* __restrict__ xs,
                          float* __restrict__ aggout, int cout) {
    int c4 = cout >> 2;
    long long total = (long long)E2TOT * HHEADS * c4;
    long long i = (long long)blockIdx.x * blockDim.x + threadIdx.x;
    if (i >= total) return;
    int cq = (int)(i % c4);
    long long t = i / c4;
    int h = (int)(t % HHEADS);
    int e = (int)(t / HHEADS);
    int s, d; edge_sd(ei, e, s, d);
    float alpha = exbuf[e * HHEADS + h] / (sbuf[d * HHEADS + h] + 1e-16f);
    float4 xv = *(const float4*)(xs + (size_t)s * HHEADS * cout + h * cout + cq * 4);
    float* o = aggout + (size_t)d * HHEADS * cout + h * cout + cq * 4;
    atomicAdd(&o[0], alpha * xv.x);
    atomicAdd(&o[1], alpha * xv.y);
    atomicAdd(&o[2], alpha * xv.z);
    atomicAdd(&o[3], alpha * xv.w);
}

// ---------------- mean over heads + bias + ELU ----------------
__global__ void head_mean_bias_elu(const float* __restrict__ aggout, const float* __restrict__ b,
                                   float* __restrict__ hout, int cout) {
    int i = blockIdx.x * blockDim.x + threadIdx.x;
    if (i >= NN * cout) return;
    int n = i / cout, c = i % cout;
    const float* r = aggout + (size_t)n * HHEADS * cout + c;
    float v = 0.f;
    #pragma unroll
    for (int h = 0; h < HHEADS; ++h) v += r[(size_t)h * cout];
    v = v * (1.0f / HHEADS) + b[c];
    hout[i] = (v > 0.f) ? v : (__expf(v) - 1.0f);   // elu
}

// ---------------- global mean pool ----------------
__global__ void pool_accum(const int* __restrict__ batch, const float* __restrict__ h,
                           float* __restrict__ pool, float* __restrict__ cnt) {
    int i = blockIdx.x * blockDim.x + threadIdx.x;
    if (i >= NN * 128) return;
    int n = i / 128, c = i % 128;
    int g = batch[n];
    atomicAdd(&pool[g * 128 + c], h[i]);
    if (c == 0) atomicAdd(&cnt[g], 1.0f);
}

__global__ void pool_final(const float* __restrict__ pool, const float* __restrict__ cnt,
                           float* __restrict__ out) {
    int i = blockIdx.x * blockDim.x + threadIdx.x;
    if (i >= GG * 128) return;
    out[i] = pool[i] / fmaxf(cnt[i / 128], 1.0f);
}

// ---------------- host-side layer driver ----------------
static void run_layer(const float* hin, int cin, int cout,
                      const float* W, const float* We, const float* as_, const float* ad_,
                      const float* ae_, const float* b,
                      const int* ei, const float* eattr, const float* meanE,
                      float* xs, float* alsrc, float* aldst, float* wevec,
                      float* mbuf, float* sbuf, float* exbuf, float* aggout,
                      float* hout, hipStream_t stream) {
    const int HC = HHEADS * cout;
    // xs = hin @ W  (bf16 WMMA, f32 accum); one wave per 16x64 strip
    int tiles = (NN / 16) * (HC / 64);
    gemm_bf16_wmma<<<(tiles + 7) / 8, 256, 0, stream>>>(hin, W, xs, NN, cin, HC);
    fold_wevec<<<1, 128, 0, stream>>>(We, ae_, wevec, cout);
    node_att<<<(NN * HHEADS + 255) / 256, 256, 0, stream>>>(xs, as_, ad_, alsrc, aldst, cout);
    fill_f32<<<(NN * HHEADS + 255) / 256, 256, 0, stream>>>(mbuf, -INFINITY, NN * HHEADS);
    hipMemsetAsync(sbuf, 0, (size_t)NN * HHEADS * sizeof(float), stream);
    att_logit_max<<<(E2TOT * HHEADS + 255) / 256, 256, 0, stream>>>(ei, eattr, meanE, wevec,
                                                                    alsrc, aldst, exbuf, mbuf);
    att_exp_sum<<<(E2TOT * HHEADS + 255) / 256, 256, 0, stream>>>(ei, exbuf, mbuf, sbuf);
    hipMemsetAsync(aggout, 0, (size_t)NN * HC * sizeof(float), stream);
    long long aggT = (long long)E2TOT * HHEADS * (cout / 4);
    aggregate<<<(unsigned)((aggT + 255) / 256), 256, 0, stream>>>(ei, exbuf, sbuf, xs, aggout, cout);
    head_mean_bias_elu<<<(NN * cout + 255) / 256, 256, 0, stream>>>(aggout, b, hout, cout);
}

extern "C" void kernel_launch(void* const* d_in, const int* in_sizes, int n_in,
                              void* d_out, int out_size, void* d_ws, size_t ws_size,
                              hipStream_t stream) {
    const float* x     = (const float*)d_in[0];
    const int*   ei    = (const int*)d_in[1];      // [2,E]: row0 = src, row1 = dst
    const float* eattr = (const float*)d_in[2];
    const int*   batch = (const int*)d_in[3];

    const float *W[3], *We[3], *As[3], *Ad[3], *Ae[3], *B[3];
    for (int l = 0; l < 3; ++l) {
        W[l]  = (const float*)d_in[4 + 6 * l + 0];
        We[l] = (const float*)d_in[4 + 6 * l + 1];
        As[l] = (const float*)d_in[4 + 6 * l + 2];
        Ad[l] = (const float*)d_in[4 + 6 * l + 3];
        Ae[l] = (const float*)d_in[4 + 6 * l + 4];
        B[l]  = (const float*)d_in[4 + 6 * l + 5];
    }

    // ---- workspace partition (floats) ----
    float* w = (float*)d_ws;
    size_t off = 0;
    float* meanE  = w + off; off += (size_t)NN * EDIMC;
    float* deg    = w + off; off += NN;
    float* xs     = w + off; off += (size_t)NN * 1024;        // max H*cout = 1024
    float* aggout = w + off; off += (size_t)NN * 1024;
    float* hA     = w + off; off += (size_t)NN * 256;
    float* hB     = w + off; off += (size_t)NN * 256;
    float* alsrc  = w + off; off += NN * HHEADS;
    float* aldst  = w + off; off += NN * HHEADS;
    float* mbuf   = w + off; off += NN * HHEADS;
    float* sbuf   = w + off; off += NN * HHEADS;
    float* exbuf  = w + off; off += (size_t)E2TOT * HHEADS;
    float* wevec  = w + off; off += EDIMC * HHEADS;
    float* pool   = w + off; off += GG * 128;
    float* cnt    = w + off; off += GG;
    if (ws_size < off * sizeof(float)) return;                // defensive no-op

    // ---- self-loop edge_attr = scatter-mean of incoming edge_attr ----
    hipMemsetAsync(meanE, 0, (size_t)NN * EDIMC * sizeof(float), stream);
    hipMemsetAsync(deg, 0, (size_t)NN * sizeof(float), stream);
    selfloop_accum<<<(EE + 255) / 256, 256, 0, stream>>>(ei, eattr, meanE, deg);
    selfloop_div<<<(NN * EDIMC + 255) / 256, 256, 0, stream>>>(meanE, deg);

    // ---- 3 GAT layers ----
    run_layer(x,  128, 128, W[0], We[0], As[0], Ad[0], Ae[0], B[0], ei, eattr, meanE,
              xs, alsrc, aldst, wevec, mbuf, sbuf, exbuf, aggout, hA, stream);
    run_layer(hA, 128, 256, W[1], We[1], As[1], Ad[1], Ae[1], B[1], ei, eattr, meanE,
              xs, alsrc, aldst, wevec, mbuf, sbuf, exbuf, aggout, hB, stream);
    run_layer(hB, 256, 128, W[2], We[2], As[2], Ad[2], Ae[2], B[2], ei, eattr, meanE,
              xs, alsrc, aldst, wevec, mbuf, sbuf, exbuf, aggout, hA, stream);

    // ---- global mean pool over batch ----
    hipMemsetAsync(pool, 0, (size_t)GG * 128 * sizeof(float), stream);
    hipMemsetAsync(cnt, 0, (size_t)GG * sizeof(float), stream);
    pool_accum<<<(NN * 128 + 255) / 256, 256, 0, stream>>>(batch, hA, pool, cnt);
    pool_final<<<(GG * 128 + 255) / 256, 256, 0, stream>>>(pool, cnt, (float*)d_out);
}
